// GATModel_30056181137901
// MI455X (gfx1250) — matrix operand order repaired and checked by reading, not compiled, not run on hardware
//
#include <hip/hip_runtime.h>
#include <hip/hip_bf16.h>
#include <math.h>

typedef unsigned short u16;
typedef __attribute__((ext_vector_type(16))) __bf16 v16bf;
typedef __attribute__((ext_vector_type(8)))  float  v8f;

// Problem constants (match reference)
#define NN    50000
#define EE    800000
#define KDIM  256        // input / hidden dim (K for both GEMMs)
#define H1    4
#define C1CH  64
#define HC1   256        // H1*C1
#define OUTD  128
#define MTILES (NN / 16) // 3125, exact

// ---------- helpers ----------
__device__ __forceinline__ u16 f32_to_bf16(float f) {
    unsigned int u = __float_as_uint(f);
    u += 0x7FFFu + ((u >> 16) & 1u);   // round to nearest even
    return (u16)(u >> 16);
}
__device__ __forceinline__ int f2ord(float f) {
    int i = __float_as_int(f);
    return (i >= 0) ? i : (i ^ 0x7FFFFFFF);   // monotonic float->int map
}
__device__ __forceinline__ float ord2f(int o) {
    int i = (o >= 0) ? o : (o ^ 0x7FFFFFFF);
    return __int_as_float(i);
}

// ---------- small utility kernels ----------
__global__ void k_f32_to_bf16(const float* __restrict__ s, u16* __restrict__ d, int n) {
    int i = blockIdx.x * blockDim.x + threadIdx.x;
    if (i < n) d[i] = f32_to_bf16(s[i]);
}
// Wt[n*K + k] = bf16(W[k*Nout + n])  (transpose so B fragments read contiguous K pairs)
__global__ void k_transpose_bf16(const float* __restrict__ W, u16* __restrict__ Wt,
                                 int K, int Nout) {
    int i = blockIdx.x * blockDim.x + threadIdx.x;
    if (i >= K * Nout) return;
    int k = i / Nout, n = i - k * Nout;
    Wt[(size_t)n * K + k] = f32_to_bf16(W[i]);
}
__global__ void k_fill_f32(float* __restrict__ p, float v, int n) {
    int i = blockIdx.x * blockDim.x + threadIdx.x;
    if (i < n) p[i] = v;
}
__global__ void k_fill_i32(int* __restrict__ p, int v, int n) {
    int i = blockIdx.x * blockDim.x + threadIdx.x;
    if (i < n) p[i] = v;
}

// ---------- WMMA GEMM: C[M,Nout] = A_bf16[M,K] * B  (B stored transposed: Bt[Nout,K]) ----------
// One wave computes a 16-row strip across all Nout columns (NT = Nout/16 tiles).
// Fragment layouts per CDNA5 ISA (16-bit A 16x32; B 32x16; f32 C/D 16x16).
template <int NT>
__global__ __launch_bounds__(128) void k_gemm_bf16(const u16* __restrict__ A,
                                                   const u16* __restrict__ Bt,
                                                   float* __restrict__ Cmat,
                                                   int Mtiles, int K, int Nout) {
    const int wib   = threadIdx.x >> 5;
    const int mtile = blockIdx.x * (blockDim.x >> 5) + wib;
    if (mtile >= Mtiles) return;           // wave-uniform exit; EXEC all-ones for WMMA
    const int lane = threadIdx.x & 31;
    const int half = lane >> 4;            // lane group 0/1
    const int l16  = lane & 15;
    const int m    = mtile * 16 + l16;     // A row for this lane

    v8f acc[NT];
#pragma unroll
    for (int t = 0; t < NT; ++t)
#pragma unroll
        for (int r = 0; r < 8; ++r) acc[t][r] = 0.0f;

    const int ktiles = K >> 5;
    for (int kt = 0; kt < ktiles; ++kt) {
        // A fragment: lane (l16, half) holds row m, K = {8*half+0..7, 16+8*half+0..7}
        union { v16bf v; unsigned int u[8]; } af;
#pragma unroll
        for (int g = 0; g < 8; ++g) {
            int kk = kt * 32 + ((g < 4) ? (8 * half + 2 * g)
                                        : (16 + 8 * half + 2 * (g - 4)));
            af.u[g] = *reinterpret_cast<const unsigned int*>(A + (size_t)m * K + kk);
        }
#pragma unroll
        for (int nt = 0; nt < NT; ++nt) {
            // B fragment: lane (l16, half) = column n, K = {8*half+0..7, 16+8*half+0..7}
            union { v16bf v; unsigned int u[8]; } bf;
            const int n = nt * 16 + l16;
#pragma unroll
            for (int g = 0; g < 8; ++g) {
                int kb = kt * 32 + ((g < 4) ? (8 * half + 2 * g)
                                            : (16 + 8 * half + 2 * (g - 4)));
                bf.u[g] = *reinterpret_cast<const unsigned int*>(Bt + (size_t)n * K + kb);
            }
            acc[nt] = __builtin_amdgcn_wmma_f32_16x16x32_bf16(
                false, af.v, false, bf.v, (short)0, acc[nt], false, false);
        }
    }
    // C/D layout: VGPR r, lanes 0-15: M=r, N=lane; lanes 16-31: M=8+r, N=lane-16
#pragma unroll
    for (int nt = 0; nt < NT; ++nt)
#pragma unroll
        for (int r = 0; r < 8; ++r) {
            int row = mtile * 16 + r + 8 * half;
            int col = nt * 16 + l16;
            Cmat[(size_t)row * Nout + col] = acc[nt][r];
        }
}

// ---------- attention coefficients: as[n,h] = sum_c h[n,h,c]*att_src[h,c] ----------
__global__ void k_alpha(const float* __restrict__ feat, const float* __restrict__ atts,
                        const float* __restrict__ attd, float* __restrict__ as_,
                        float* __restrict__ ad_, int Nn, int H, int C) {
    int i = blockIdx.x * blockDim.x + threadIdx.x;
    if (i >= Nn * H) return;
    int n = i / H, h = i - n * H;
    const float* fp = feat + (size_t)n * H * C + (size_t)h * C;
    float s = 0.f, d = 0.f;
    for (int c = 0; c < C; ++c) {
        float v = fp[c];
        s += v * atts[h * C + c];
        d += v * attd[h * C + c];
    }
    as_[i] = s;
    ad_[i] = d;
}

// ---------- edge pass 1: segment max via ordered-int atomicMax ----------
__global__ void k_edge_max(const int* __restrict__ ei, const float* __restrict__ as_,
                           const float* __restrict__ ad_, int* __restrict__ emax,
                           int E, int Nn, int H) {
    int i = blockIdx.x * blockDim.x + threadIdx.x;
    int ET = E + Nn;
    if (i >= ET * H) return;
    int e = i / H, h = i - e * H;
    int s, d;
    if (e < E) { s = ei[e]; d = ei[E + e]; } else { s = e - E; d = s; }
    float t = as_[s * H + h] + ad_[d * H + h];
    t = (t > 0.f) ? t : 0.2f * t;          // leaky_relu(0.2)
    atomicMax(&emax[d * H + h], f2ord(t));
}

// ---------- edge pass 2: accumulate exp(e-emax) and exp(e-emax)*h[src] ----------
// One wave32 per edge; lane handles channels lane, lane+32, ...
__global__ void k_edge_accum(const int* __restrict__ ei, const float* __restrict__ as_,
                             const float* __restrict__ ad_, const int* __restrict__ emax,
                             const float* __restrict__ feat, float* __restrict__ denom,
                             float* __restrict__ accum, int E, int Nn, int H, int C) {
    int gtid = blockIdx.x * blockDim.x + threadIdx.x;
    int wave = gtid >> 5;
    int lane = gtid & 31;
    int ET = E + Nn;
    if (wave >= ET) return;
    int s, d;
    if (wave < E) { s = ei[wave]; d = ei[E + wave]; } else { s = wave - E; d = s; }
    const int HC = H * C;
    float ex[4];
    for (int h = 0; h < H; ++h) {
        float t = as_[s * H + h] + ad_[d * H + h];
        t = (t > 0.f) ? t : 0.2f * t;
        ex[h] = expf(t - ord2f(emax[d * H + h]));
    }
    if (lane < H) atomicAdd(&denom[d * H + lane], ex[lane]);
    const float* fp = feat + (size_t)s * HC;
    float* ap = accum + (size_t)d * HC;
    for (int c = lane; c < HC; c += 32) {
        int h = c / C;
        atomicAdd(&ap[c], ex[h] * fp[c]);
    }
}

// ---------- layer-1 finish: divide, bias, exact GELU, quantize to bf16 ----------
__global__ void k_finish1(const float* __restrict__ accum, const float* __restrict__ denom,
                          const float* __restrict__ bias, u16* __restrict__ hmid) {
    int i = blockIdx.x * blockDim.x + threadIdx.x;
    if (i >= NN * HC1) return;
    int n = i >> 8, c = i & 255, h = c >> 6;
    float v = accum[i] / (denom[n * H1 + h] + 1e-16f) + bias[c];
    float g = 0.5f * v * (1.0f + erff(v * 0.70710678118654752f));  // exact-erf GELU
    hmid[i] = f32_to_bf16(g);
}

// ---------- layer-2 finish (in place on d_out): divide + bias ----------
__global__ void k_finish2(float* __restrict__ out, const float* __restrict__ denom,
                          const float* __restrict__ bias) {
    int i = blockIdx.x * blockDim.x + threadIdx.x;
    if (i >= NN * OUTD) return;
    int n = i >> 7, c = i & 127;
    out[i] = out[i] / (denom[n] + 1e-16f) + bias[c];
}

// =====================================================================
extern "C" void kernel_launch(void* const* d_in, const int* in_sizes, int n_in,
                              void* d_out, int out_size, void* d_ws, size_t ws_size,
                              hipStream_t stream) {
    const float* x    = (const float*)d_in[0];
    const int*   ei   = (const int*)d_in[1];     // [2,E] src row then dst row
    const float* W1   = (const float*)d_in[2];   // [256,256]
    const float* aS1  = (const float*)d_in[3];   // [4,64]
    const float* aD1  = (const float*)d_in[4];
    const float* b1   = (const float*)d_in[5];   // [256]
    const float* W2   = (const float*)d_in[6];   // [256,128]
    const float* aS2  = (const float*)d_in[7];   // [1,128]
    const float* aD2  = (const float*)d_in[8];
    const float* b2   = (const float*)d_in[9];   // [128]
    float* out = (float*)d_out;                  // [50000,128]

    // workspace carve-out (256B aligned regions)
    size_t off = 0;
    char* base = (char*)d_ws;
    auto alloc = [&](size_t bytes) -> void* {
        void* p = base + off;
        off += (bytes + 255) & ~(size_t)255;
        return p;
    };
    u16*   xbf   = (u16*)  alloc((size_t)NN * KDIM * 2);   // x bf16; reused as hmid bf16
    u16*   w1t   = (u16*)  alloc((size_t)KDIM * HC1 * 2);  // W1^T bf16 [256,256]
    u16*   w2t   = (u16*)  alloc((size_t)OUTD * KDIM * 2); // W2^T bf16 [128,256]
    float* h1    = (float*)alloc((size_t)NN * HC1 * 4);    // h1 f32; reused as h2
    float* as_   = (float*)alloc((size_t)NN * H1 * 4);
    float* ad_   = (float*)alloc((size_t)NN * H1 * 4);
    int*   emax  = (int*)  alloc((size_t)NN * H1 * 4);
    float* denom = (float*)alloc((size_t)NN * H1 * 4);
    float* acc1  = (float*)alloc((size_t)NN * HC1 * 4);

    const int TB = 256;
    auto cdiv = [](long long a, long long b) { return (unsigned)((a + b - 1) / b); };

    // ---- precision staging ----
    k_f32_to_bf16<<<cdiv((long long)NN * KDIM, TB), TB, 0, stream>>>(x, xbf, NN * KDIM);
    k_transpose_bf16<<<cdiv(KDIM * HC1, TB), TB, 0, stream>>>(W1, w1t, KDIM, HC1);
    k_transpose_bf16<<<cdiv(KDIM * OUTD, TB), TB, 0, stream>>>(W2, w2t, KDIM, OUTD);

    // ---- layer 1 ----
    k_gemm_bf16<16><<<cdiv(MTILES, 4), 128, 0, stream>>>(xbf, w1t, h1, MTILES, KDIM, HC1);
    k_alpha<<<cdiv(NN * H1, TB), TB, 0, stream>>>(h1, aS1, aD1, as_, ad_, NN, H1, C1CH);
    k_fill_f32<<<cdiv(NN * H1, TB), TB, 0, stream>>>(denom, 0.f, NN * H1);
    k_fill_i32<<<cdiv(NN * H1, TB), TB, 0, stream>>>(emax, (int)0x80000000, NN * H1);
    k_fill_f32<<<cdiv((long long)NN * HC1, TB), TB, 0, stream>>>(acc1, 0.f, NN * HC1);
    k_edge_max<<<cdiv((long long)(EE + NN) * H1, TB), TB, 0, stream>>>(ei, as_, ad_, emax, EE, NN, H1);
    k_edge_accum<<<cdiv((long long)(EE + NN) * 32, TB), TB, 0, stream>>>(
        ei, as_, ad_, emax, h1, denom, acc1, EE, NN, H1, C1CH);
    k_finish1<<<cdiv((long long)NN * HC1, TB), TB, 0, stream>>>(acc1, denom, b1, xbf);

    // ---- layer 2 (hmid = xbf region, h2 = h1 region, accum = d_out) ----
    k_gemm_bf16<8><<<cdiv(MTILES, 4), 128, 0, stream>>>(xbf, w2t, h1, MTILES, KDIM, OUTD);
    k_alpha<<<cdiv(NN, TB), TB, 0, stream>>>(h1, aS2, aD2, as_, ad_, NN, 1, OUTD);
    k_fill_f32<<<cdiv(NN, TB), TB, 0, stream>>>(denom, 0.f, NN);
    k_fill_i32<<<cdiv(NN, TB), TB, 0, stream>>>(emax, (int)0x80000000, NN);
    k_fill_f32<<<cdiv((long long)NN * OUTD, TB), TB, 0, stream>>>(out, 0.f, NN * OUTD);
    k_edge_max<<<cdiv((long long)(EE + NN), TB), TB, 0, stream>>>(ei, as_, ad_, emax, EE, NN, 1);
    k_edge_accum<<<cdiv((long long)(EE + NN) * 32, TB), TB, 0, stream>>>(
        ei, as_, ad_, emax, h1, denom, out, EE, NN, 1, OUTD);
    k_finish2<<<cdiv((long long)NN * OUTD, TB), TB, 0, stream>>>(out, denom, b2);
}
